// MaskedMultiHeadAttn_12309376270872
// MI455X (gfx1250) — matrix operand order repaired and checked by reading, not compiled, not run on hardware
//
#include <hip/hip_runtime.h>
#include <hip/hip_bf16.h>
#include <stdint.h>

typedef __attribute__((ext_vector_type(16))) _Float16 v16h;
typedef __attribute__((ext_vector_type(8)))  _Float16 v8h;
typedef __attribute__((ext_vector_type(8)))  float    v8f;

#define WMMA_F16(a, b, c) \
  __builtin_amdgcn_wmma_f32_16x16x32_f16(false, (a), false, (b), (short)0, (c), false, false)

// Build a v16h fragment from two 16-byte LDS chunks (two ds_load_b128).
static __device__ __forceinline__ v16h make_frag(const _Float16* p0, const _Float16* p1) {
  v8h lo = *(const v8h*)p0;
  v8h hi = *(const v8h*)p1;
  v16h r;
#pragma unroll
  for (int i = 0; i < 8; ++i) { r[i] = lo[i]; r[i + 8] = hi[i]; }
  return r;
}

// gfx1250 async copy global -> LDS, 16 bytes per lane, tracked by ASYNCcnt.
// LDS offset = low 32 bits of the generic pointer (local offset in low dword).
static __device__ __forceinline__ void async_ld16(_Float16* lds_dst, const _Float16* g_src) {
  unsigned int l = (unsigned int)(uintptr_t)lds_dst;
  unsigned long long g = (unsigned long long)(uintptr_t)g_src;
  asm volatile("global_load_async_to_lds_b128 %0, %1, off" :: "v"(l), "v"(g) : "memory");
}
static __device__ __forceinline__ void wait_async0() {
  asm volatile("s_wait_asynccnt 0" ::: "memory");
}

// ---------------------------------------------------------------- convert f32 -> f16
__global__ __launch_bounds__(256)
void cvt_f32_f16(const float* __restrict__ in, _Float16* __restrict__ out, int n) {
  int i = (blockIdx.x * 256 + threadIdx.x) * 4;
  if (i + 3 < n) {
    float4 v = *(const float4*)(in + i);
    out[i + 0] = (_Float16)v.x;
    out[i + 1] = (_Float16)v.y;
    out[i + 2] = (_Float16)v.z;
    out[i + 3] = (_Float16)v.w;
  }
}

// -------------------------------------------------- convert + transpose W (KxN f32 -> NxK f16)
__global__ __launch_bounds__(256)
void cvt_transpose_w(const float* __restrict__ in, _Float16* __restrict__ out) {
  __shared__ float t[32 * 33];
  const int N = 1024, K = 1024;
  int k0 = blockIdx.y * 32, n0 = blockIdx.x * 32;
  int tx = threadIdx.x & 31, ty = threadIdx.x >> 5;
#pragma unroll
  for (int r = ty; r < 32; r += 8) t[r * 33 + tx] = in[(size_t)(k0 + r) * N + n0 + tx];
  __syncthreads();
#pragma unroll
  for (int r = ty; r < 32; r += 8) out[(size_t)(n0 + r) * K + k0 + tx] = (_Float16)t[tx * 33 + r];
}

// -------------------------------------------------- transpose V per (b,h): [s][64] -> [64][2048]
__global__ __launch_bounds__(256)
void transpose_v(const _Float16* __restrict__ Vh, _Float16* __restrict__ VhT) {
  __shared__ __align__(16) _Float16 t[64 * 72];
  int bh = blockIdx.y, b = bh >> 4, h = bh & 15;
  int j0 = blockIdx.x * 64;
  int r = threadIdx.x >> 2, c0 = (threadIdx.x & 3) * 16;
  const _Float16* src = Vh + ((size_t)(b * 2048 + j0 + r)) * 1024 + h * 64 + c0;
  *(v8h*)&t[r * 72 + c0]     = *(const v8h*)src;
  *(v8h*)&t[r * 72 + c0 + 8] = *(const v8h*)(src + 8);
  __syncthreads();
  int d = threadIdx.x >> 2, jc = (threadIdx.x & 3) * 16;
  v8h o0, o1;
#pragma unroll
  for (int e = 0; e < 8; ++e) {
    o0[e] = t[(jc + e) * 72 + d];
    o1[e] = t[(jc + 8 + e) * 72 + d];
  }
  _Float16* dst = VhT + ((size_t)bh * 64 + d) * 2048 + j0 + jc;
  *(v8h*)dst       = o0;
  *(v8h*)(dst + 8) = o1;
}

// ---------------------------------------------------------------- WMMA GEMM
// C[M,N] = A[M,K] @ W[K,N] + bias, with W pre-transposed (Wt is N x K).
// Block = 256 thr (8 waves), tile 128x64, KT=32, async double-buffered LDS.
template <int OUT16>
__global__ __launch_bounds__(256)
void gemm16(const _Float16* __restrict__ A, const _Float16* __restrict__ Wt,
            const float* __restrict__ bias, void* __restrict__ Cout,
            int M, int N, int K) {
  // layout (f16): A buffers 2 x 128*40, then B buffers 2 x 64*40
  __shared__ __align__(16) _Float16 smem[2 * 128 * 40 + 2 * 64 * 40];
  const int ABUF = 128 * 40, BBASE = 2 * 128 * 40, BBUF = 64 * 40;

  const int tid  = threadIdx.x;
  const int lane = tid & 31, wave = tid >> 5;
  const int lh = lane >> 4, ll = lane & 15;
  const int m0 = blockIdx.y * 128, n0 = blockIdx.x * 64;

  auto fillA = [&](int buf, int k0) {
    int r = tid >> 1, hc = (tid & 1) * 16;
    const _Float16* src = A + (size_t)(m0 + r) * K + k0 + hc;
    _Float16* dst = smem + buf * ABUF + r * 40 + hc;
    async_ld16(dst,     src);
    async_ld16(dst + 8, src + 8);
  };
  auto fillB = [&](int buf, int k0) {
    int n = tid >> 2, kc = (tid & 3) * 8;
    async_ld16(smem + BBASE + buf * BBUF + n * 40 + kc,
               Wt + (size_t)(n0 + n) * K + k0 + kc);
  };

  v8f acc[4] = {};
  fillA(0, 0);
  fillB(0, 0);
  int cur = 0;

  for (int k0 = 0; k0 < K; k0 += 32) {
    wait_async0();
    __syncthreads();
    if (k0 + 32 < K) { fillA(cur ^ 1, k0 + 32); fillB(cur ^ 1, k0 + 32); }

    const _Float16* Ac = smem + cur * ABUF;
    const _Float16* Bc = smem + BBASE + cur * BBUF;
    const int arow = wave * 16 + ll;
    v16h afrag = make_frag(&Ac[arow * 40 + lh * 8], &Ac[arow * 40 + lh * 8 + 16]);
    v16h bfr[4];
#pragma unroll
    for (int jj = 0; jj < 4; ++jj) {
      int n = jj * 16 + ll;
      bfr[jj] = make_frag(&Bc[n * 40 + lh * 8], &Bc[n * 40 + lh * 8 + 16]);
    }
#pragma unroll
    for (int jj = 0; jj < 4; ++jj) acc[jj] = WMMA_F16(afrag, bfr[jj], acc[jj]);
    cur ^= 1;
  }

#pragma unroll
  for (int jj = 0; jj < 4; ++jj) {
    int col = n0 + jj * 16 + ll;
    float bv = bias[col];
#pragma unroll
    for (int e = 0; e < 8; ++e) {
      int row = m0 + wave * 16 + lh * 8 + e;
      float v = acc[jj][e] + bv;
      if (OUT16) ((_Float16*)Cout)[(size_t)row * N + col] = (_Float16)v;
      else       ((float*)Cout)[(size_t)row * N + col]    = v;
    }
  }
}

// ---------------------------------------------------------------- flash attention
// scores[i,j] = K[i]·Q[j] (no scale), causal j<=i, softmax over j, out = P @ V.
// Block: 256 thr (8 waves), one (b,h), i-tile = 128 (16 rows/wave).
// Q/V tiles async into double buffers; V pre-transposed (VhT) so its tiles are
// contiguous async copies too. amask converted once to an LDS f16 bias.
__global__ __launch_bounds__(256)
void attn_kernel(const _Float16* __restrict__ Qh, const _Float16* __restrict__ Kh,
                 const _Float16* __restrict__ VhT, const int* __restrict__ amask,
                 _Float16* __restrict__ ctx) {
  // pool (f16): Kt/P 9216 | Qt x2 9216 | Vt x2 9216 | amask bias 2048
  __shared__ __align__(16) _Float16 smem[29696];
  const int QBASE = 9216, VBASE = 18432, TBUF = 4608, ABASE = 27648;
  _Float16* Kt  = smem;          // 128 x 72; reused as P staging
  _Float16* Amb = smem + ABASE;  // 2048 f16 additive bias

  const int tid  = threadIdx.x;
  const int lane = tid & 31, wave = tid >> 5;
  const int lh = lane >> 4, ll = lane & 15;
  const int i0 = blockIdx.x * 128;
  const int bh = blockIdx.y;
  const int b = bh >> 4, h = bh & 15;
  const size_t rowbase = (size_t)b * 2048;
  const int hcol = h * 64;
  const float NEG_INF = -__builtin_inff();

  // amask -> additive f16 bias in LDS (once per block)
  for (int j = tid; j < 2048; j += 256)
    Amb[j] = amask[b * 2048 + j] ? (_Float16)0.f : (_Float16)NEG_INF;

  {  // K i-tile 128 x 64, async
    int r = tid >> 1, c0 = (tid & 1) * 32;
    const _Float16* src = Kh + (rowbase + i0 + r) * 1024 + hcol + c0;
#pragma unroll
    for (int q = 0; q < 4; ++q) async_ld16(&Kt[r * 72 + c0 + q * 8], src + q * 8);
  }
  wait_async0();
  __syncthreads();

  // K fragments (A-matrix 16x32 per d-half) kept in registers for all tiles
  v16h kfrag[2];
  {
    int r = wave * 16 + ll;
#pragma unroll
    for (int m = 0; m < 2; ++m)
      kfrag[m] = make_frag(&Kt[r * 72 + m * 32 + lh * 8],
                           &Kt[r * 72 + m * 32 + lh * 8 + 16]);
  }

  auto fillQ = [&](int buf, int jt) {
    int r = tid >> 2, c0 = (tid & 3) * 16;
    const _Float16* src = Qh + (rowbase + jt + r) * 1024 + hcol + c0;
    _Float16* dst = smem + QBASE + buf * TBUF + r * 72 + c0;
    async_ld16(dst,     src);
    async_ld16(dst + 8, src + 8);
  };
  auto fillV = [&](int buf, int jt) {
    int d = tid >> 2, jc = (tid & 3) * 16;
    const _Float16* src = VhT + ((size_t)bh * 64 + d) * 2048 + jt + jc;
    _Float16* dst = smem + VBASE + buf * TBUF + d * 72 + jc;
    async_ld16(dst,     src);
    async_ld16(dst + 8, src + 8);
  };

  float mrow[8], lrow[8];
  v8f oacc[4] = {};
#pragma unroll
  for (int e = 0; e < 8; ++e) { mrow[e] = NEG_INF; lrow[e] = 0.f; }

  const int ntiles = (i0 >> 6) + 2;  // 64-wide j-tiles through the diagonal
  _Float16* Pw = &Kt[wave * 16 * 72];
  fillQ(0, 0);
  fillV(0, 0);
  int cur = 0;

  for (int t = 0; t < ntiles; ++t) {
    const int jt = t * 64;
    wait_async0();
    __syncthreads();
    if (t + 1 < ntiles) { fillQ(cur ^ 1, jt + 64); fillV(cur ^ 1, jt + 64); }
    const _Float16* Qt = smem + QBASE + cur * TBUF;
    const _Float16* Vt = smem + VBASE + cur * TBUF;

    // scores: load 4 B-fragments, then 4 WMMAs, per d-half
    v8f sacc[4];
    {
      v16h bf[4];
#pragma unroll
      for (int jj = 0; jj < 4; ++jj) {
        int n = jj * 16 + ll;
        bf[jj] = make_frag(&Qt[n * 72 + lh * 8], &Qt[n * 72 + lh * 8 + 16]);
      }
#pragma unroll
      for (int jj = 0; jj < 4; ++jj) { v8f z = {}; sacc[jj] = WMMA_F16(kfrag[0], bf[jj], z); }
#pragma unroll
      for (int jj = 0; jj < 4; ++jj) {
        int n = jj * 16 + ll;
        bf[jj] = make_frag(&Qt[n * 72 + 32 + lh * 8], &Qt[n * 72 + 32 + lh * 8 + 16]);
      }
#pragma unroll
      for (int jj = 0; jj < 4; ++jj) sacc[jj] = WMMA_F16(kfrag[1], bf[jj], sacc[jj]);
    }

    // mask (LDS bias + causal) and online softmax
    float p[4][8], tmax[8];
#pragma unroll
    for (int e = 0; e < 8; ++e) tmax[e] = NEG_INF;
#pragma unroll
    for (int jj = 0; jj < 4; ++jj) {
      int jg = jt + jj * 16 + ll;
      float mb = (float)Amb[jg];
#pragma unroll
      for (int e = 0; e < 8; ++e) {
        int ig = i0 + wave * 16 + lh * 8 + e;
        float s = (jg <= ig) ? (sacc[jj][e] + mb) : NEG_INF;
        p[jj][e] = s;
        tmax[e] = fmaxf(tmax[e], s);
      }
    }
#pragma unroll
    for (int e = 0; e < 8; ++e)
#pragma unroll
      for (int m = 1; m <= 8; m <<= 1)
        tmax[e] = fmaxf(tmax[e], __shfl_xor(tmax[e], m, 32));

    float alpha[8], rsum[8];
#pragma unroll
    for (int e = 0; e < 8; ++e) {
      float mn = fmaxf(mrow[e], tmax[e]);
      alpha[e] = __expf(mrow[e] - mn);
      mrow[e] = mn;
      float s = 0.f;
#pragma unroll
      for (int jj = 0; jj < 4; ++jj) {
        float pe = __expf(p[jj][e] - mn);
        p[jj][e] = pe;
        s += pe;
      }
      rsum[e] = s;
    }
#pragma unroll
    for (int e = 0; e < 8; ++e) {
#pragma unroll
      for (int m = 1; m <= 8; m <<= 1)
        rsum[e] += __shfl_xor(rsum[e], m, 32);
      lrow[e] = lrow[e] * alpha[e] + rsum[e];
    }
#pragma unroll
    for (int dd = 0; dd < 4; ++dd)
#pragma unroll
      for (int e = 0; e < 8; ++e)
        oacc[dd][e] *= alpha[e];

    // stage P (C-layout -> row-major LDS) so it can be re-read in A-layout
#pragma unroll
    for (int jj = 0; jj < 4; ++jj)
#pragma unroll
      for (int e = 0; e < 8; ++e)
        Pw[(lh * 8 + e) * 72 + jj * 16 + ll] = (_Float16)p[jj][e];

    // out += P(16x64) @ V(64x64): per j-half, 1 A-frag + 4 B-frags then 4 WMMAs
#pragma unroll
    for (int jh = 0; jh < 2; ++jh) {
      v16h pfrag = make_frag(&Pw[ll * 72 + jh * 32 + lh * 8],
                             &Pw[ll * 72 + jh * 32 + lh * 8 + 16]);
      v16h vf[4];
#pragma unroll
      for (int dd = 0; dd < 4; ++dd) {
        int n = dd * 16 + ll;
        vf[dd] = make_frag(&Vt[n * 72 + jh * 32 + lh * 8],
                           &Vt[n * 72 + jh * 32 + lh * 8 + 16]);
      }
#pragma unroll
      for (int dd = 0; dd < 4; ++dd) oacc[dd] = WMMA_F16(pfrag, vf[dd], oacc[dd]);
    }
    cur ^= 1;
  }

#pragma unroll
  for (int e = 0; e < 8; ++e) lrow[e] = 1.f / lrow[e];
#pragma unroll
  for (int dd = 0; dd < 4; ++dd)
#pragma unroll
    for (int e = 0; e < 8; ++e) {
      int ig = i0 + wave * 16 + lh * 8 + e;
      ctx[(rowbase + ig) * 1024 + hcol + dd * 16 + ll] = (_Float16)(oacc[dd][e] * lrow[e]);
    }
}

// ---------------------------------------------------------------- launch
extern "C" void kernel_launch(void* const* d_in, const int* in_sizes, int n_in,
                              void* d_out, int out_size, void* d_ws, size_t ws_size,
                              hipStream_t stream) {
  const float* x     = (const float*)d_in[0];
  const int*   amask = (const int*)d_in[1];
  const float* Wq = (const float*)d_in[2];
  const float* bq = (const float*)d_in[3];
  const float* Wk = (const float*)d_in[4];
  const float* bk = (const float*)d_in[5];
  const float* Wv = (const float*)d_in[6];
  const float* bv = (const float*)d_in[7];
  const float* Wo = (const float*)d_in[8];
  const float* bo = (const float*)d_in[9];
  float* out = (float*)d_out;

  const int M = 4096, D = 1024;  // M = B*S

  _Float16* ws  = (_Float16*)d_ws;
  _Float16* xh  = ws; ws += (size_t)M * D;
  _Float16* WqT = ws; ws += (size_t)D * D;
  _Float16* WkT = ws; ws += (size_t)D * D;
  _Float16* WvT = ws; ws += (size_t)D * D;
  _Float16* WoT = ws; ws += (size_t)D * D;
  _Float16* Qh  = ws; ws += (size_t)M * D;
  _Float16* Kh  = ws; ws += (size_t)M * D;
  _Float16* Vh  = ws; ws += (size_t)M * D;
  _Float16* VhT = ws; ws += (size_t)M * D;
  _Float16* Ctx = ws; ws += (size_t)M * D;

  cvt_f32_f16<<<(M * D) / 1024, 256, 0, stream>>>(x, xh, M * D);
  dim3 gt(D / 32, D / 32);
  cvt_transpose_w<<<gt, 256, 0, stream>>>(Wq, WqT);
  cvt_transpose_w<<<gt, 256, 0, stream>>>(Wk, WkT);
  cvt_transpose_w<<<gt, 256, 0, stream>>>(Wv, WvT);
  cvt_transpose_w<<<gt, 256, 0, stream>>>(Wo, WoT);

  dim3 gg(D / 64, M / 128);
  gemm16<1><<<gg, 256, 0, stream>>>(xh, WqT, bq, Qh, M, D, D);
  gemm16<1><<<gg, 256, 0, stream>>>(xh, WkT, bk, Kh, M, D, D);
  gemm16<1><<<gg, 256, 0, stream>>>(xh, WvT, bv, Vh, M, D, D);

  transpose_v<<<dim3(2048 / 64, 32), 256, 0, stream>>>(Vh, VhT);

  attn_kernel<<<dim3(2048 / 128, 32), 256, 0, stream>>>(Qh, Kh, VhT, amask, Ctx);

  gemm16<0><<<gg, 256, 0, stream>>>(Ctx, WoT, bo, out, M, D, D);
}
